// GCN_40140764348986
// MI455X (gfx1250) — matrix operand order repaired and checked by reading, not compiled, not run on hardware
//
#include <hip/hip_runtime.h>

#define N_NODES  50000
#define N_EDGES  800000
#define N_GRAPHS 256
#define IN_CH    128
#define HID      256

typedef __attribute__((ext_vector_type(16))) __bf16 v16bf;
typedef __attribute__((ext_vector_type(8)))  float  v8f;

// ---------- helpers ----------
__device__ __forceinline__ __bf16 f2bf(float f) {
  union { float f; unsigned u; } x; x.f = f;
  unsigned r = x.u + 0x7FFFu + ((x.u >> 16) & 1u);   // round-to-nearest-even
  unsigned short hs = (unsigned short)(r >> 16);
  __bf16 h; __builtin_memcpy(&h, &hs, sizeof(h));
  return h;
}
__device__ __forceinline__ float bf2f(__bf16 b) {
  unsigned short hs; __builtin_memcpy(&hs, &b, sizeof(hs));
  union { unsigned u; float f; } x; x.u = ((unsigned)hs) << 16;
  return x.f;
}
// Hardware f32 atomic add (no return -> STOREcnt; s_endpgm implies wait-idle).
__device__ __forceinline__ void atomAdd(float* p, float v) {
  asm volatile("global_atomic_add_f32 %0, %1, off scope:SCOPE_DEV"
               :: "v"(p), "v"(v) : "memory");
}

// ---------- small elementwise kernels ----------
__global__ void fill_k(float* p, float v, int n) {
  int i = blockIdx.x * blockDim.x + threadIdx.x;
  if (i < n) p[i] = v;
}
__global__ void deg_count_k(const int* __restrict__ col, float* __restrict__ deg) {
  int e = blockIdx.x * blockDim.x + threadIdx.x;
  if (e < N_EDGES) atomAdd(&deg[col[e]], 1.0f);
}
__global__ void rsqrt_k(float* d) {
  int i = blockIdx.x * blockDim.x + threadIdx.x;
  if (i < N_NODES) d[i] = rsqrtf(d[i]);   // deg >= 1 always (self loop)
}
__global__ void norm_k(const int* __restrict__ row, const int* __restrict__ col,
                       const float* __restrict__ dis, float* __restrict__ nrm) {
  int e = blockIdx.x * blockDim.x + threadIdx.x;
  if (e < N_EDGES) nrm[e] = dis[row[e]] * dis[col[e]];
}

// ---------- weight pre-pack into WMMA B-fragment order ----------
// Packed layout: [kc][tile][lane][i], flat t = ((kc*16 + tile)*32 + lane)*16 + i
//   k = kc*32 + 16*(lane>>4) + i ,  n = tile*16 + (lane&15)
// so each lane's 16 bf16 B-fragment elements are 32 contiguous bytes.
__global__ void pack_w_k(const float* __restrict__ w, __bf16* __restrict__ hiP,
                         __bf16* __restrict__ loP, int K) {
  int t = blockIdx.x * blockDim.x + threadIdx.x;   // K*256 elements
  if (t >= K * 256) return;
  int i    = t & 15;
  int lane = (t >> 4) & 31;
  int tile = (t >> 9) & 15;
  int kc   = t >> 13;
  int k = kc * 32 + ((lane >> 4) * 16) + i;
  int n = tile * 16 + (lane & 15);
  float f = w[(size_t)k * 256 + n];
  __bf16 h = f2bf(f);
  hiP[t] = h;
  loP[t] = f2bf(f - bf2f(h));
}

// ---------- GEMM: C[M x 256] = A[M x K] @ W[K x 256], W fragment-packed hi/lo bf16 ----------
// 256 threads = 8 waves; wave (w&1) picks a 16-row strip, (w>>1) picks a 64-col group.
// Precision: 3-product bf16 split (A_hi*B_hi + A_hi*B_lo + A_lo*B_hi) ~= 16 mantissa bits.
__global__ void __launch_bounds__(256) gemm_wmma_k(
    const float* __restrict__ A, const v16bf* __restrict__ BhiP,
    const v16bf* __restrict__ BloP, float* __restrict__ C, int M, int K)
{
  const int lane = threadIdx.x & 31;
  const int wave = threadIdx.x >> 5;
  const int half = lane >> 4;       // K-half selector (this code's contiguous k = 16h+s order)
  const int ln   = lane & 15;
  const int rowTile = ((int)blockIdx.x * 2 + (wave & 1)) * 16;
  const int colBase = (wave >> 1) * 64;
  const int tile0   = colBase >> 4;

  int arow = rowTile + ln;
  if (arow > M - 1) arow = M - 1;   // clamp loads; stores are predicated
  const float* Ap = A + (size_t)arow * K;

  v8f acc[4] = {};

  for (int k0 = 0; k0 < K; k0 += 32) {
    // A fragment (16x32 bf16): two contiguous 8-float runs per lane
    float av[16];
    {
      const float4* p0 = (const float4*)(Ap + k0 + 16 * half);       // k = k0+16h+[0,8)
      const float4* p1 = (const float4*)(Ap + k0 + 16 * half + 8);   // k = k0+16h+8+[0,8)
      ((float4*)av)[0] = p0[0];
      ((float4*)av)[1] = p0[1];
      ((float4*)av)[2] = p1[0];
      ((float4*)av)[3] = p1[1];
    }
    v16bf ahi, alo;
#pragma unroll
    for (int i = 0; i < 16; ++i) {
      __bf16 h = f2bf(av[i]);
      ahi[i] = h;
      alo[i] = f2bf(av[i] - bf2f(h));
    }

    const size_t fragBase = ((size_t)(k0 >> 5) * 16 + tile0) * 32 + lane;
#pragma unroll
    for (int t = 0; t < 4; ++t) {
      const v16bf bhi = BhiP[fragBase + (size_t)t * 32];   // 32B contiguous per lane
      const v16bf blo = BloP[fragBase + (size_t)t * 32];
      acc[t] = __builtin_amdgcn_wmma_f32_16x16x32_bf16(
          false, ahi, false, bhi, (short)0, acc[t], false, false);
      acc[t] = __builtin_amdgcn_wmma_f32_16x16x32_bf16(
          false, ahi, false, blo, (short)0, acc[t], false, false);
      acc[t] = __builtin_amdgcn_wmma_f32_16x16x32_bf16(
          false, alo, false, bhi, (short)0, acc[t], false, false);
    }
  }

  // C/D layout: VGPR r -> M = r + 8*half, N = ln
#pragma unroll
  for (int t = 0; t < 4; ++t) {
#pragma unroll
    for (int r = 0; r < 8; ++r) {
      int m = rowTile + r + 8 * half;
      if (m < M) C[(size_t)m * 256 + colBase + t * 16 + ln] = acc[t][r];
    }
  }
}

// ---------- aggregation ----------
// out = xw * dis^2 (self-loop) + bias   (also serves as the zero-init for atomics)
__global__ void agg_init_k(const float4* __restrict__ xw, const float* __restrict__ dis,
                           const float* __restrict__ bias, float4* __restrict__ out) {
  int t = blockIdx.x * blockDim.x + threadIdx.x;   // N_NODES*64 float4's
  if (t >= N_NODES * 64) return;
  int i = t >> 6, c4 = t & 63;
  float s = dis[i]; s *= s;
  float4 v = xw[t];
  float4 b = ((const float4*)bias)[c4];
  float4 o;
  o.x = v.x * s + b.x; o.y = v.y * s + b.y;
  o.z = v.z * s + b.z; o.w = v.w * s + b.w;
  out[t] = o;
}

// one wave per edge; 32 lanes x float4 x 2 iters = 256 channels
__global__ void agg_edges_k(const float4* __restrict__ xw,
                            const int* __restrict__ row, const int* __restrict__ col,
                            const float* __restrict__ nrm, float* __restrict__ out) {
  int gid = blockIdx.x * blockDim.x + threadIdx.x;
  int e = gid >> 5, lane = gid & 31;
  if (e >= N_EDGES) return;
  int r = row[e], c = col[e];
  float w = nrm[e];
  const float4* src = xw + (size_t)r * 64;
  float* dst = out + (size_t)c * 256;
#pragma unroll
  for (int it = 0; it < 2; ++it) {
    int ci = lane + it * 32;
    float4 v = src[ci];
    atomAdd(dst + ci * 4 + 0, v.x * w);
    atomAdd(dst + ci * 4 + 1, v.y * w);
    atomAdd(dst + ci * 4 + 2, v.z * w);
    atomAdd(dst + ci * 4 + 3, v.w * w);
  }
}

__global__ void relu_k(float4* p, int n4) {
  int t = blockIdx.x * blockDim.x + threadIdx.x;
  if (t >= n4) return;
  float4 v = p[t];
  v.x = fmaxf(v.x, 0.f); v.y = fmaxf(v.y, 0.f);
  v.z = fmaxf(v.z, 0.f); v.w = fmaxf(v.w, 0.f);
  p[t] = v;
}

// ---------- pooling ----------
__global__ void pool_cnt_k(const int* __restrict__ bat, float* __restrict__ cnt) {
  int i = blockIdx.x * blockDim.x + threadIdx.x;
  if (i < N_NODES) atomAdd(&cnt[bat[i]], 1.0f);
}
__global__ void pool_sum_k(const float4* __restrict__ h, const int* __restrict__ bat,
                           float* __restrict__ pooled) {
  int t = blockIdx.x * blockDim.x + threadIdx.x;   // N_NODES*64
  if (t >= N_NODES * 64) return;
  int i = t >> 6, c4 = t & 63;
  int g = bat[i];
  float4 v = h[t];
  float* dst = pooled + (size_t)g * 256 + c4 * 4;
  atomAdd(dst + 0, v.x); atomAdd(dst + 1, v.y);
  atomAdd(dst + 2, v.z); atomAdd(dst + 3, v.w);
}
__global__ void pool_div_k(float* __restrict__ pooled, const float* __restrict__ cnt) {
  int t = blockIdx.x * blockDim.x + threadIdx.x;   // 256*256
  if (t >= N_GRAPHS * 256) return;
  pooled[t] /= fmaxf(cnt[t >> 8], 1.0f);
}

// ---------- FFN tail ----------
__global__ void bias_relu_k(float* __restrict__ g, const float* __restrict__ b) {
  int t = blockIdx.x * blockDim.x + threadIdx.x;   // 256*256
  if (t >= N_GRAPHS * 256) return;
  g[t] = fmaxf(g[t] + b[t & 255], 0.0f);
}
__global__ void ffn_final_k(const float* __restrict__ g, const float* __restrict__ Wo,
                            const float* __restrict__ bo, float* __restrict__ out) {
  int gi = blockIdx.x * blockDim.x + threadIdx.x;
  if (gi >= N_GRAPHS) return;
  float s = bo[0];
  const float* gp = g + (size_t)gi * 256;
#pragma unroll 8
  for (int c = 0; c < 256; ++c) s += gp[c] * Wo[c];
  out[gi] = s;
}

// ---------- driver ----------
extern "C" void kernel_launch(void* const* d_in, const int* in_sizes, int n_in,
                              void* d_out, int out_size, void* d_ws, size_t ws_size,
                              hipStream_t stream) {
  const float* x   = (const float*)d_in[0];
  const int*   ei  = (const int*)d_in[1];
  const int*   bat = (const int*)d_in[2];
  const float* W1  = (const float*)d_in[3];
  const float* b1  = (const float*)d_in[4];
  const float* W2  = (const float*)d_in[5];
  const float* b2  = (const float*)d_in[6];
  const float* W3  = (const float*)d_in[7];
  const float* b3  = (const float*)d_in[8];
  const float* Wf  = (const float*)d_in[9];
  const float* bfv = (const float*)d_in[10];
  const float* Wo  = (const float*)d_in[11];
  const float* bo  = (const float*)d_in[12];
  float* out = (float*)d_out;

  const int* row = ei;
  const int* col = ei + N_EDGES;

  char* ws = (char*)d_ws;
  float*  dis    = (float*)(ws + 0);           // N          (deg -> rsqrt in place)
  float*  nrm    = (float*)(ws + 200192);      // E
  float*  bufA   = (float*)(ws + 3400192);     // N*256 (xw)
  float*  bufB   = (float*)(ws + 54600192);    // N*256 (h)
  __bf16* whi    = (__bf16*)(ws + 105800192);  // 229376 bf16 (fragment-packed)
  __bf16* wlo    = (__bf16*)(ws + 106258944);  // 229376 bf16
  float*  pooled = (float*)(ws + 106717696);   // 256*256
  float*  gbuf   = (float*)(ws + 106979840);   // 256*256
  float*  cnt    = (float*)(ws + 107241984);   // 256
  (void)in_sizes; (void)n_in; (void)out_size; (void)ws_size;

  // bf16 hi/lo weight splits, packed back to back
  __bf16* w1hi = whi;            __bf16* w1lo = wlo;            // 128*256
  __bf16* w2hi = whi + 32768;    __bf16* w2lo = wlo + 32768;    // 256*256
  __bf16* w3hi = whi + 98304;    __bf16* w3lo = wlo + 98304;
  __bf16* wfhi = whi + 163840;   __bf16* wflo = wlo + 163840;

  const int BT = 256;
  const int gN    = (N_NODES + BT - 1) / BT;
  const int gE    = (N_EDGES + BT - 1) / BT;
  const int gNC4  = (N_NODES * 64) / BT;       // 12500 exact
  const int gEdgW = (N_EDGES * 32) / BT;       // 100000 exact (1 wave / edge)
  const int gGEMM = (N_NODES + 31) / 32;       // 1563
  const int gPool = (N_GRAPHS * 256) / BT;     // 256

  // ---- normalization: deg = 1 (self loop) + in-degree; dis = rsqrt(deg); norm = dis[r]*dis[c]
  fill_k<<<gN, BT, 0, stream>>>(dis, 1.0f, N_NODES);
  deg_count_k<<<gE, BT, 0, stream>>>(col, dis);
  rsqrt_k<<<gN, BT, 0, stream>>>(dis);
  norm_k<<<gE, BT, 0, stream>>>(row, col, dis, nrm);

  // ---- weight splits (fragment-packed); grid = K blocks of 256 (K*256 elems)
  pack_w_k<<<IN_CH, BT, 0, stream>>>(W1, w1hi, w1lo, IN_CH);
  pack_w_k<<<HID,   BT, 0, stream>>>(W2, w2hi, w2lo, HID);
  pack_w_k<<<HID,   BT, 0, stream>>>(W3, w3hi, w3lo, HID);
  pack_w_k<<<HID,   BT, 0, stream>>>(Wf, wfhi, wflo, HID);

  // ---- layer 1: x(50000x128) @ W1 -> bufA ; aggregate -> bufB ; relu
  gemm_wmma_k<<<gGEMM, BT, 0, stream>>>(x, (const v16bf*)w1hi, (const v16bf*)w1lo,
                                        bufA, N_NODES, IN_CH);
  agg_init_k<<<gNC4, BT, 0, stream>>>((const float4*)bufA, dis, b1, (float4*)bufB);
  agg_edges_k<<<gEdgW, BT, 0, stream>>>((const float4*)bufA, row, col, nrm, bufB);
  relu_k<<<gNC4, BT, 0, stream>>>((float4*)bufB, N_NODES * 64);

  // ---- layer 2
  gemm_wmma_k<<<gGEMM, BT, 0, stream>>>(bufB, (const v16bf*)w2hi, (const v16bf*)w2lo,
                                        bufA, N_NODES, HID);
  agg_init_k<<<gNC4, BT, 0, stream>>>((const float4*)bufA, dis, b2, (float4*)bufB);
  agg_edges_k<<<gEdgW, BT, 0, stream>>>((const float4*)bufA, row, col, nrm, bufB);
  relu_k<<<gNC4, BT, 0, stream>>>((float4*)bufB, N_NODES * 64);

  // ---- layer 3
  gemm_wmma_k<<<gGEMM, BT, 0, stream>>>(bufB, (const v16bf*)w3hi, (const v16bf*)w3lo,
                                        bufA, N_NODES, HID);
  agg_init_k<<<gNC4, BT, 0, stream>>>((const float4*)bufA, dis, b3, (float4*)bufB);
  agg_edges_k<<<gEdgW, BT, 0, stream>>>((const float4*)bufA, row, col, nrm, bufB);
  relu_k<<<gNC4, BT, 0, stream>>>((float4*)bufB, N_NODES * 64);

  // ---- global mean pool
  fill_k<<<1, BT, 0, stream>>>(cnt, 0.0f, N_GRAPHS);
  fill_k<<<gPool, BT, 0, stream>>>(pooled, 0.0f, N_GRAPHS * 256);
  pool_cnt_k<<<gN, BT, 0, stream>>>(bat, cnt);
  pool_sum_k<<<gNC4, BT, 0, stream>>>((const float4*)bufB, bat, pooled);
  pool_div_k<<<gPool, BT, 0, stream>>>(pooled, cnt);

  // ---- FFN: relu(pooled @ Wf + bf) @ Wo + bo
  gemm_wmma_k<<<N_GRAPHS / 32, BT, 0, stream>>>(pooled, (const v16bf*)wfhi,
                                                (const v16bf*)wflo, gbuf, N_GRAPHS, HID);
  bias_relu_k<<<gPool, BT, 0, stream>>>(gbuf, bfv);
  ffn_final_k<<<1, BT, 0, stream>>>(gbuf, Wo, bo, out);
}
